// ScaledDotProductAttention_37125697307285
// MI455X (gfx1250) — matrix operand order repaired and checked by reading, not compiled, not run on hardware
//
#include <hip/hip_runtime.h>
#include <stdint.h>

// ---------------------------------------------------------------------------
// Scaled-dot-product attention (exp -> mask-to-zero -> normalize), B=16,
// S=2048, D=64, fp32 in/out.
//   - matmuls: v_wmma_f32_16x16x32_bf16 (f32 accumulate)
//   - mask (67 MB, dominant traffic): TDM tensor_load_to_lds, double-buffered,
//     tracked with TENSORcnt
//   - softmax scale folded into Q as 0.125*log2(e); exp via hardware exp2
// ---------------------------------------------------------------------------

typedef __attribute__((ext_vector_type(16))) __bf16   v16bf;
typedef __attribute__((ext_vector_type(8)))  float    v8f;
typedef __attribute__((ext_vector_type(4)))  uint32_t u32x4;
typedef __attribute__((ext_vector_type(8)))  uint32_t u32x8;

#define SLEN   2048
#define DHEAD  64
#define WAVES  8
#define KCHUNK 32                  // keys per chunk (= PV contraction)
#define QROWS  (WAVES * 16)        // 128 Q rows per block

struct US16 { unsigned short s[16]; };
struct US8  { unsigned short s[8];  };
struct U4x2 { uint4 a, b; };

// f32 -> bf16 round-to-nearest-even (used on the cold staging path)
static __device__ __forceinline__ unsigned short f2bf(float f) {
  uint32_t u = __builtin_bit_cast(uint32_t, f);
  uint32_t r = u + 0x7FFFu + ((u >> 16) & 1u);
  return (unsigned short)(r >> 16);
}

// f32 -> bf16 round-half-up (hot path: P staging; 2 VALU ops)
static __device__ __forceinline__ unsigned short f2bf_fast(float f) {
  uint32_t u = __builtin_bit_cast(uint32_t, f);
  return (unsigned short)((u + 0x8000u) >> 16);
}

static __device__ __forceinline__ float fast_exp2(float x) {
#if __has_builtin(__builtin_amdgcn_exp2f)
  return __builtin_amdgcn_exp2f(x);   // v_exp_f32
#else
  return exp2f(x);
#endif
}

// A-fragment (16xK, K=32, bf16) per ISA 7.12.2:
// dwords 0-3: K = 8h + {0..7}; dwords 4-7: K = 16 + 8h + {0..7}
static __device__ __forceinline__ v16bf fragA_lds(const unsigned short* row, int h) {
  U4x2 t;
  t.a = *(const uint4*)(row + 8 * h);
  t.b = *(const uint4*)(row + 16 + 8 * h);
  return __builtin_bit_cast(v16bf, t);
}

static __device__ __forceinline__ v16bf fragA_gmem_f32_scaled(const float* row, int dbase,
                                                              int h, float scale) {
  US16 t;
#pragma unroll
  for (int i = 0; i < 8; ++i) t.s[i]     = f2bf(row[dbase + 8 * h + i] * scale);
#pragma unroll
  for (int i = 0; i < 8; ++i) t.s[8 + i] = f2bf(row[dbase + 16 + 8 * h + i] * scale);
  return __builtin_bit_cast(v16bf, t);
}

// B-fragment (KxN, K=32, N=16, bf16): lane half h holds K = 8h + {0..7} then
// K = 16 + 8h + {0..7} for its column, 2 packed per dword (same order as A).
static __device__ __forceinline__ v16bf fragB_lds(const unsigned short* col, int h) {
  U4x2 t;
  t.a = *(const uint4*)(col + 8 * h);
  t.b = *(const uint4*)(col + 16 + 8 * h);
  return __builtin_bit_cast(v16bf, t);
}

// TDM: DMA a 2D mask tile (KCHUNK bytes x QROWS rows, row stride SLEN) from
// global memory into LDS at lds_off. Descriptor per ISA ch.8 (groups 0,1 only
// => up to 2D tensor). Issued by one wave; completion tracked with TENSORcnt.
static __device__ __forceinline__ void tdm_load_mask_tile(const unsigned char* gsrc,
                                                          uint32_t lds_off) {
  const uint64_t ga = (uint64_t)(uintptr_t)gsrc;
  u32x4 g0;
  g0[0] = 1u;                                                  // count=1, user mode
  g0[1] = lds_off;                                             // lds_addr[31:0]
  g0[2] = (uint32_t)ga;                                        // global_addr[31:0]
  g0[3] = (uint32_t)((ga >> 32) & 0x01FFFFFFu) | 0x80000000u;  // addr[56:32] | type=2
  u32x8 g1;
  g1[0] = 0u;                                   // wg_mask=0, data_size=1B, no flags
  g1[1] = ((uint32_t)SLEN & 0xFFFFu) << 16;     // tensor_dim0[15:0] = 2048
  g1[2] = ((uint32_t)SLEN & 0xFFFFu) << 16;     // dim0 hi=0 | tensor_dim1[15:0] = 2048
  g1[3] = (uint32_t)KCHUNK << 16;               // dim1 hi=0 | tile_dim0 = 32 bytes
  g1[4] = (uint32_t)QROWS;                      // tile_dim1 = 128 rows, tile_dim2 = 0
  g1[5] = (uint32_t)SLEN;                       // tensor_dim0_stride = 2048 (low 32)
  g1[6] = 0u;                                   // stride hi | dim1_stride lo
  g1[7] = 0u;
  asm volatile("tensor_load_to_lds %0, %1" :: "s"(g0), "s"(g1) : "memory");
}

__global__ __launch_bounds__(256, 1)
void sdpa_wmma_kernel(const float* __restrict__ Q, const float* __restrict__ K,
                      const float* __restrict__ V, const unsigned char* __restrict__ M,
                      float* __restrict__ O)
{
  __shared__ unsigned short Ks[KCHUNK * DHEAD];          // [key][d]  bf16 (4 KB)
  __shared__ unsigned short Vt[DHEAD * KCHUNK];          // [d][key]  bf16 (4 KB)
  __shared__ unsigned short Pst[WAVES][16 * KCHUNK];     // per-wave P staging (8 KB)
  __shared__ unsigned char  Msk[2][QROWS * KCHUNK];      // double-buffered mask (8 KB)

  const int tid  = threadIdx.x;
  const int wave = tid >> 5;
  const int lane = tid & 31;
  const int h    = lane >> 4;   // lane half (wave32)
  const int n    = lane & 15;   // column / row-in-tile index

  const int b     = blockIdx.x >> 4;
  const int tile  = blockIdx.x & 15;
  const int q0blk = tile * QROWS;           // block's first Q row
  const int q0    = q0blk + wave * 16;      // wave's first Q row

  const float* Qb = Q + (size_t)b * SLEN * DHEAD;
  const float* Kb = K + (size_t)b * SLEN * DHEAD;
  const float* Vb = V + (size_t)b * SLEN * DHEAD;
  const unsigned char* Mb = M + (size_t)b * SLEN * SLEN + (size_t)q0blk * SLEN;
  float* Ob = O + (size_t)b * SLEN * DHEAD;

  // Q tile A-fragments with softmax scale folded in: exp(s/8) == exp2(s * log2e/8)
  const float SCALE = 0.18033688011112042f;   // log2(e) / sqrt(64)
  const float* qrow = Qb + (size_t)(q0 + n) * DHEAD;
  const v16bf qf0 = fragA_gmem_f32_scaled(qrow, 0,  h, SCALE);
  const v16bf qf1 = fragA_gmem_f32_scaled(qrow, 32, h, SCALE);

  v8f acc0 = {}, acc1 = {}, acc2 = {}, acc3 = {};
  float rs[8];
#pragma unroll
  for (int r = 0; r < 8; ++r) rs[r] = 0.0f;

  const int srow = tid >> 3;        // staged key row (0..31)
  const int scol = (tid & 7) * 8;   // staged d-slice (8 floats)

  // Prime the mask pipeline: DMA chunk 0 into buffer 0.
  if (wave == 0) {
    tdm_load_mask_tile(Mb, (uint32_t)(uintptr_t)&Msk[0][0]);
  }

  for (int kt = 0; kt < SLEN; kt += KCHUNK) {
    const int cur = (kt >> 5) & 1;

    __syncthreads();   // all fragment/mask reads of the previous chunk are done

    // Kick off the NEXT chunk's mask DMA into the other buffer (overlaps with
    // this chunk's staging + compute).
    if (wave == 0 && (kt + KCHUNK) < SLEN) {
      tdm_load_mask_tile(Mb + (size_t)(kt + KCHUNK), (uint32_t)(uintptr_t)&Msk[cur ^ 1][0]);
    }

    // ---- cooperative stage: K chunk -> Ks (bf16), V chunk -> Vt (bf16, transposed)
    {
      const float* kr = Kb + (size_t)(kt + srow) * DHEAD + scol;
      const float* vr = Vb + (size_t)(kt + srow) * DHEAD + scol;
      float4 a = ((const float4*)kr)[0];
      float4 c = ((const float4*)kr)[1];
      US8 t;
      t.s[0] = f2bf(a.x); t.s[1] = f2bf(a.y); t.s[2] = f2bf(a.z); t.s[3] = f2bf(a.w);
      t.s[4] = f2bf(c.x); t.s[5] = f2bf(c.y); t.s[6] = f2bf(c.z); t.s[7] = f2bf(c.w);
      *(uint4*)(Ks + srow * DHEAD + scol) = __builtin_bit_cast(uint4, t);

      float4 va = ((const float4*)vr)[0];
      float4 vc = ((const float4*)vr)[1];
      unsigned short vs[8] = { f2bf(va.x), f2bf(va.y), f2bf(va.z), f2bf(va.w),
                               f2bf(vc.x), f2bf(vc.y), f2bf(vc.z), f2bf(vc.w) };
#pragma unroll
      for (int i = 0; i < 8; ++i) Vt[(scol + i) * KCHUNK + srow] = vs[i];

      if (kt + KCHUNK < SLEN) {   // gfx1250 global_prefetch_b8 for next K/V chunk
        __builtin_prefetch(kr + KCHUNK * DHEAD, 0, 1);
        __builtin_prefetch(vr + KCHUNK * DHEAD, 0, 1);
      }
    }

    // This chunk's mask DMA (issued last iteration) must have landed. In steady
    // state one newer DMA is in flight (in-order TENSORcnt => wait <= 1); on the
    // last chunk nothing new was issued (wait 0).
    if (wave == 0) {
      if (kt + KCHUNK < SLEN) __builtin_amdgcn_s_wait_tensorcnt(1);
      else                    __builtin_amdgcn_s_wait_tensorcnt(0);
    }
    __syncthreads();

    // ---- S = Q * K^T for keys [kt..kt+15] (s0) and [kt+16..kt+31] (s1)
    v8f s0 = {}, s1 = {};
    {
      v16bf kf;
      kf = fragB_lds(Ks + (size_t)n * DHEAD, h);
      s0 = __builtin_amdgcn_wmma_f32_16x16x32_bf16(false, qf0, false, kf, (short)0, s0, false, false);
      kf = fragB_lds(Ks + (size_t)n * DHEAD + 32, h);
      s0 = __builtin_amdgcn_wmma_f32_16x16x32_bf16(false, qf1, false, kf, (short)0, s0, false, false);
      kf = fragB_lds(Ks + (size_t)(16 + n) * DHEAD, h);
      s1 = __builtin_amdgcn_wmma_f32_16x16x32_bf16(false, qf0, false, kf, (short)0, s1, false, false);
      kf = fragB_lds(Ks + (size_t)(16 + n) * DHEAD + 32, h);
      s1 = __builtin_amdgcn_wmma_f32_16x16x32_bf16(false, qf1, false, kf, (short)0, s1, false, false);
    }

    // ---- exp2 (scale pre-folded), mask-to-zero, row-sum; stage P (bf16).
    // C-layout: vgpr r -> row m = r + 8h, col = n.
    const unsigned char* mrow = &Msk[cur][(wave * 16 + 8 * h) * KCHUNK + n];
#pragma unroll
    for (int r = 0; r < 8; ++r) {
      const unsigned char* mp = mrow + r * KCHUNK;
      float e0 = fast_exp2(s0[r]);
      float e1 = fast_exp2(s1[r]);
      e0 = mp[0]  ? 0.0f : e0;
      e1 = mp[16] ? 0.0f : e1;
      rs[r] += e0 + e1;
      Pst[wave][(r + 8 * h) * KCHUNK + n]      = f2bf_fast(e0);
      Pst[wave][(r + 8 * h) * KCHUNK + 16 + n] = f2bf_fast(e1);
    }

    // intra-wave LDS RAW fence before reading P back in A-layout
    __asm__ __volatile__("s_wait_dscnt 0" ::: "memory");

    // ---- acc += P(16x32) * V(32x64), one WMMA per 16-wide d group
    const v16bf pf = fragA_lds(&Pst[wave][(size_t)n * KCHUNK], h);
    v16bf vf;
    vf = fragB_lds(Vt + (size_t)(0 * 16 + n) * KCHUNK, h);
    acc0 = __builtin_amdgcn_wmma_f32_16x16x32_bf16(false, pf, false, vf, (short)0, acc0, false, false);
    vf = fragB_lds(Vt + (size_t)(1 * 16 + n) * KCHUNK, h);
    acc1 = __builtin_amdgcn_wmma_f32_16x16x32_bf16(false, pf, false, vf, (short)0, acc1, false, false);
    vf = fragB_lds(Vt + (size_t)(2 * 16 + n) * KCHUNK, h);
    acc2 = __builtin_amdgcn_wmma_f32_16x16x32_bf16(false, pf, false, vf, (short)0, acc2, false, false);
    vf = fragB_lds(Vt + (size_t)(3 * 16 + n) * KCHUNK, h);
    acc3 = __builtin_amdgcn_wmma_f32_16x16x32_bf16(false, pf, false, vf, (short)0, acc3, false, false);
  }

  // ---- reduce denominators across the 16-lane half (wave32 shuffles)
#pragma unroll
  for (int r = 0; r < 8; ++r) {
    float v = rs[r];
    v += __shfl_xor(v, 1, 32);
    v += __shfl_xor(v, 2, 32);
    v += __shfl_xor(v, 4, 32);
    v += __shfl_xor(v, 8, 32);
    rs[r] = 1.0f / v;
  }

  // ---- out[q0 + r + 8h][g*16 + n] = acc_g[r] * (1/rowsum)
  float* orow = Ob + (size_t)(q0 + 8 * h) * DHEAD + n;
#pragma unroll
  for (int r = 0; r < 8; ++r) {
    float* op = orow + (size_t)r * DHEAD;
    op[0]  = acc0[r] * rs[r];
    op[16] = acc1[r] * rs[r];
    op[32] = acc2[r] * rs[r];
    op[48] = acc3[r] * rs[r];
  }
}

extern "C" void kernel_launch(void* const* d_in, const int* in_sizes, int n_in,
                              void* d_out, int out_size, void* d_ws, size_t ws_size,
                              hipStream_t stream) {
  (void)in_sizes; (void)n_in; (void)out_size; (void)d_ws; (void)ws_size;
  const float* q = (const float*)d_in[0];
  const float* k = (const float*)d_in[1];
  const float* v = (const float*)d_in[2];
  const unsigned char* m = (const unsigned char*)d_in[3];  // jax bool -> 1 byte/elem
  float* out = (float*)d_out;

  // 16 batches x 16 row-tiles (128 Q rows per block, 8 waves x 16 rows)
  dim3 grid(16 * (SLEN / QROWS));
  sdpa_wmma_kernel<<<grid, 256, 0, stream>>>(q, k, v, m, out);
}
